// GatedDeltaProduct_85864986181836
// MI455X (gfx1250) — matrix operand order repaired and checked by reading, not compiled, not run on hardware
//
#include <hip/hip_runtime.h>
#include <math.h>

// ---------------------------------------------------------------------------
// Types
// ---------------------------------------------------------------------------
typedef __bf16 bf16_t;
typedef __attribute__((ext_vector_type(16))) __bf16 v16bf;
typedef __attribute__((ext_vector_type(8)))  float  v8f;

#define DEV_INLINE __device__ __forceinline__

union PK8 {   // 8 x bf16 = 16 bytes
    uint4 u;
    bf16_t h[8];
};
union PK16 {  // 16 x bf16 = 32 bytes (one WMMA operand fragment)
    uint4 u[2];
    v16bf v;
};

// ---------------------------------------------------------------------------
// Problem constants
// ---------------------------------------------------------------------------
static constexpr int Bsz  = 2;
static constexpr int Tlen = 1024;
static constexpr int HIDn = 2048;
static constexpr int NHp  = 2;
static constexpr int Hh   = 6;
static constexpr int DKc  = 256;
static constexpr int DVc  = 512;
static constexpr int KEYD = Hh * DKc;     // 1536
static constexpr int VALD = Hh * DVc;     // 3072
static constexpr int TI   = Tlen * NHp;   // 2048 (interleaved length)
static constexpr int CHKl = 64;
static constexpr int NCHUNK = TI / CHKl;  // 32
static constexpr int DVS  = 128;          // DV slice per workgroup
static constexpr int NSL  = DVc / DVS;    // 4

// ---------------------------------------------------------------------------
// CDNA5 async global->LDS copy (16B per lane), tracked by ASYNCcnt.
// ---------------------------------------------------------------------------
DEV_INLINE void async_b128(void* lds, const void* gaddr) {
    unsigned l = (unsigned)(uintptr_t)lds;
    asm volatile("global_load_async_to_lds_b128 %0, %1, off"
                 :: "v"(l), "v"(gaddr) : "memory");
}
DEV_INLINE void wait_async0() {
    asm volatile("s_wait_asynccnt 0x0" ::: "memory");
}

// ---------------------------------------------------------------------------
// WMMA helpers (wave32, contiguous k = 16*half + s in both A and B), fully vectorized:
//   A frag: row-major [M][K] tile -> 2x b128 loads per lane
//   B frag: N-major   [N][K] tile -> 2x b128 loads per lane
// ---------------------------------------------------------------------------
DEV_INLINE v8f v8f_zero() {
    v8f z = {0.f, 0.f, 0.f, 0.f, 0.f, 0.f, 0.f, 0.f};
    return z;
}

DEV_INLINE v8f wmma_bf16(v16bf a, v16bf b, v8f c) {
    return __builtin_amdgcn_wmma_f32_16x16x32_bf16(
        false, a, false, b, (short)0, c, false, false);
}

DEV_INLINE v16bf frag_a_vec(const bf16_t* p, int mbase, int kbase, int ld) {
    int lane = (int)(threadIdx.x & 31);
    int half = lane >> 4;
    const bf16_t* base = p + (mbase + (lane & 15)) * ld + kbase + half * 16;
    PK16 t;
    t.u[0] = *(const uint4*)(base);
    t.u[1] = *(const uint4*)(base + 8);
    return t.v;
}

DEV_INLINE v16bf frag_b_vec(const bf16_t* p, int kbase, int nbase, int ld) {
    int lane = (int)(threadIdx.x & 31);
    int half = lane >> 4;
    const bf16_t* base = p + (nbase + (lane & 15)) * ld + kbase + half * 16;
    PK16 t;
    t.u[0] = *(const uint4*)(base);
    t.u[1] = *(const uint4*)(base + 8);
    return t.v;
}

// Same as frag_b_vec but with size_t leading dim (direct-from-global weights).
DEV_INLINE v16bf frag_b_glb(const bf16_t* p, size_t ld, int nbase, int kbase) {
    int lane = (int)(threadIdx.x & 31);
    int half = lane >> 4;
    const bf16_t* base = p + (size_t)(nbase + (lane & 15)) * ld + kbase + half * 16;
    PK16 t;
    t.u[0] = *(const uint4*)(base);
    t.u[1] = *(const uint4*)(base + 8);
    return t.v;
}

// C/D tile (M = half*8 + r contiguous) from an N-major bf16 tile.
DEV_INLINE v8f load_cT(const bf16_t* p, int mbase, int nbase, int ld) {
    int lane = (int)(threadIdx.x & 31);
    int half = lane >> 4;
    PK8 t;
    t.u = *(const uint4*)(p + (nbase + (lane & 15)) * ld + mbase + half * 8);
    v8f c;
#pragma unroll
    for (int r = 0; r < 8; ++r) c[r] = (float)t.h[r];
    return c;
}

DEV_INLINE void store_cT(bf16_t* p, int mbase, int nbase, int ld, v8f c,
                         float scale) {
    int lane = (int)(threadIdx.x & 31);
    int half = lane >> 4;
    PK8 t;
#pragma unroll
    for (int r = 0; r < 8; ++r) t.h[r] = (bf16_t)(c[r] * scale);
    *(uint4*)(p + (nbase + (lane & 15)) * ld + mbase + half * 8) = t.u;
}

// ---------------------------------------------------------------------------
// Cast kernels: plain f32->bf16, and f32[K][N] -> bf16[N][K] (weight transpose)
// ---------------------------------------------------------------------------
__global__ void cast_f32_to_bf16(const float* __restrict__ s,
                                 bf16_t* __restrict__ d, long n) {
    long i = (long)blockIdx.x * blockDim.x + threadIdx.x;
    long stride = (long)gridDim.x * blockDim.x;
    for (; i < n; i += stride) d[i] = (bf16_t)s[i];
}

__global__ void cast_transpose_f32_to_bf16(const float* __restrict__ s,
                                           bf16_t* __restrict__ d,
                                           int K, int N) {
    long total = (long)K * N;
    long i = (long)blockIdx.x * blockDim.x + threadIdx.x;
    long stride = (long)gridDim.x * blockDim.x;
    for (; i < total; i += stride) {
        int k = (int)(i / N);
        int n = (int)(i - (long)k * N);
        d[(size_t)n * K + k] = (bf16_t)s[i];
    }
}

// ---------------------------------------------------------------------------
// bf16 WMMA GEMM: C[M,N] (f32) = A[M,K] @ BT[N,K]^T.
//   A tile staged into LDS with global_load_async_to_lds_b128 (ASYNCcnt path).
//   B fragments stream directly from the pre-transposed, L2-resident weights
//   as 2x global_load_b128 per fragment (no LDS staging / scatter).
// Block: 256 threads (8 waves), tile 128x128, K-step 32.
// ---------------------------------------------------------------------------
__global__ void __launch_bounds__(256)
gemm_bf16(const bf16_t* __restrict__ A, const bf16_t* __restrict__ BT,
          float* __restrict__ C, int M, int N, int K) {
    __shared__ bf16_t As[128 * 32];    // row-major [m][k]

    int bm = blockIdx.y * 128;
    int bn = blockIdx.x * 128;
    int tid = (int)threadIdx.x;
    int lane = tid & 31;
    int w = tid >> 5;
    int wm = (w >> 1) * 32;
    int wn = (w & 1) * 64;

    v8f acc[2][4];
#pragma unroll
    for (int i = 0; i < 2; ++i)
#pragma unroll
        for (int j = 0; j < 4; ++j) acc[i][j] = v8f_zero();

    int arow = tid >> 1;
    int acol = (tid & 1) * 16;
    const bf16_t* aRow = A + (size_t)(bm + arow) * K + acol;
    bf16_t* ldsA = As + arow * 32 + acol;

    for (int k0 = 0; k0 < K; k0 += 32) {
        // async-copy A tile 128x32 into LDS (2 x 16B per thread)
        const bf16_t* ga = aRow + k0;
        async_b128(ldsA, ga);
        async_b128(ldsA + 8, ga + 8);

        // B fragments straight from global (pre-transposed weights in L2)
        v16bf bfr[4];
#pragma unroll
        for (int j = 0; j < 4; ++j)
            bfr[j] = frag_b_glb(BT, (size_t)K, bn + wn + j * 16, k0);

        if (k0 + 32 < K)  // global_prefetch_b8 for next B lines
            __builtin_prefetch(BT + (size_t)(bn + wn + (lane & 15)) * K + k0 + 32,
                               0, 0);

        wait_async0();
        __syncthreads();

        v16bf af[2];
#pragma unroll
        for (int i = 0; i < 2; ++i)
            af[i] = frag_a_vec(As, wm + i * 16, 0, 32);
#pragma unroll
        for (int i = 0; i < 2; ++i)
#pragma unroll
            for (int j = 0; j < 4; ++j)
                acc[i][j] = wmma_bf16(af[i], bfr[j], acc[i][j]);
        __syncthreads();
    }

    // epilogue: one row pointer per lane, stepped by N, immediate col offsets
    int half = lane >> 4, nn = lane & 15;
#pragma unroll
    for (int i = 0; i < 2; ++i) {
        float* crow =
            C + (size_t)(bm + wm + i * 16 + half * 8) * N + (bn + wn + nn);
#pragma unroll
        for (int r = 0; r < 8; ++r) {
#pragma unroll
            for (int j = 0; j < 4; ++j) crow[j * 16] = acc[i][j][r];
            crow += N;
        }
    }
}

// ---------------------------------------------------------------------------
// conv(K=4, causal) + SiLU + per-head L2 norm, write interleaved bf16 q/k.
// ---------------------------------------------------------------------------
__global__ void __launch_bounds__(256)
conv_norm_qk(const float* __restrict__ src,   // [B, T, KEYD]
             const float* __restrict__ cw,    // [KEYD, 4]
             bf16_t* __restrict__ dst,        // [B, TI, H, DK]
             int sub, float outscale) {
    int blk = (int)blockIdx.x;
    int h = blk % Hh;
    int t = (blk / Hh) % Tlen;
    int b = blk / (Hh * Tlen);
    int tid = (int)threadIdx.x;
    int ch = h * DKc + tid;

    float y = 0.f;
#pragma unroll
    for (int j = 0; j < 4; ++j) {
        int ts = t - 3 + j;
        if (ts >= 0)
            y += src[(size_t)(b * Tlen + ts) * KEYD + ch] * cw[ch * 4 + j];
    }
    y = y / (1.f + expf(-y));  // SiLU

    __shared__ float red[256];
    red[tid] = y * y;
    __syncthreads();
#pragma unroll
    for (int s = 128; s > 0; s >>= 1) {
        if (tid < s) red[tid] += red[tid + s];
        __syncthreads();
    }
    float rn = rsqrtf(red[0] + 1e-12f);

    size_t o = (((size_t)b * TI + (t * NHp + sub)) * Hh + h) * DKc + tid;
    dst[o] = (bf16_t)(y * rn * outscale);
}

// conv + SiLU for v, write interleaved bf16.
__global__ void __launch_bounds__(256)
conv_silu_v(const float* __restrict__ src,   // [B, T, VALD]
            const float* __restrict__ cw,    // [VALD, 4]
            bf16_t* __restrict__ dst,        // [B, TI, H*DV]
            int sub) {
    int blk = (int)blockIdx.x;
    int t = blk % Tlen;
    int b = blk / Tlen;
    int tid = (int)threadIdx.x;
    size_t drow = ((size_t)b * TI + (t * NHp + sub)) * VALD;
#pragma unroll
    for (int it = 0; it < VALD / 256; ++it) {
        int ch = tid + it * 256;
        float y = 0.f;
#pragma unroll
        for (int j = 0; j < 4; ++j) {
            int ts = t - 3 + j;
            if (ts >= 0)
                y += src[(size_t)(b * Tlen + ts) * VALD + ch] * cw[ch * 4 + j];
        }
        y = y / (1.f + expf(-y));
        dst[drow + ch] = (bf16_t)y;
    }
}

// beta = sigmoid(x @ b_ws), g = -exp(A_log)*softplus(x @ a_w + dt_bias)
__global__ void __launch_bounds__(256)
beta_g_kernel(const float* __restrict__ x,
              const float* __restrict__ bws,
              const float* __restrict__ aw,
              const float* __restrict__ Alog,
              const float* __restrict__ dtb,
              float* __restrict__ bOut,   // [B, TI, H]
              float* __restrict__ gOut) { // [B, TI, H]
    int blk = (int)blockIdx.x;
    int t = blk % Tlen;
    int b = blk / Tlen;
    int tid = (int)threadIdx.x;
    const float* xr = x + (size_t)(b * Tlen + t) * HIDn;
    __shared__ float red[256];

    for (int sub = 0; sub < NHp; ++sub) {
        for (int h = 0; h < Hh; ++h) {
            float p = 0.f;
            for (int c = tid; c < HIDn; c += 256)
                p += xr[c] * bws[(size_t)sub * HIDn * Hh + (size_t)c * Hh + h];
            red[tid] = p;
            __syncthreads();
#pragma unroll
            for (int s = 128; s > 0; s >>= 1) {
                if (tid < s) red[tid] += red[tid + s];
                __syncthreads();
            }
            if (tid == 0)
                bOut[((size_t)b * TI + (t * NHp + sub)) * Hh + h] =
                    1.f / (1.f + expf(-red[0]));
            __syncthreads();
        }
    }
    for (int h = 0; h < Hh; ++h) {
        float p = 0.f;
        for (int c = tid; c < HIDn; c += 256)
            p += xr[c] * aw[(size_t)c * Hh + h];
        red[tid] = p;
        __syncthreads();
#pragma unroll
        for (int s = 128; s > 0; s >>= 1) {
            if (tid < s) red[tid] += red[tid + s];
            __syncthreads();
        }
        if (tid == 0) {
            float z = red[0] + dtb[h];
            float sp = (z > 20.f) ? z : log1pf(expf(z));
            float g = -expf(Alog[h]) * sp;
            gOut[((size_t)b * TI + (t * NHp + 0)) * Hh + h] = g;
#pragma unroll
            for (int s2 = 1; s2 < NHp; ++s2)
                gOut[((size_t)b * TI + (t * NHp + s2)) * Hh + h] = 0.f;
        }
        __syncthreads();
    }
}

// ---------------------------------------------------------------------------
// Chunked gated delta rule (WY form). One block per (b, h, dv-slice).
// K/Q chunk tiles staged with async global->LDS copies; all transposes are
// 8x8 register-block transposes; every WMMA fragment is 2x ds_load_b128.
// ---------------------------------------------------------------------------
static constexpr size_t OFF_ST  = 0;                  // S^T   [128][256] bf16
static constexpr size_t OFF_K   = OFF_ST  + 65536;    // K     [64][256]  bf16
static constexpr size_t OFF_Q   = OFF_K   + 32768;    // Q     [64][256]  bf16
static constexpr size_t OFF_KI  = OFF_Q   + 32768;    // Ki    [64][256]  bf16
static constexpr size_t OFF_KIT = OFF_KI  + 32768;    // Ki^T  [256][64]  bf16
static constexpr size_t OFF_VT  = OFF_KIT + 32768;    // V^T   [128][64]  bf16
static constexpr size_t OFF_UT  = OFF_VT  + 16384;    // U^T   [128][64]  f32
static constexpr size_t OFF_UBT = OFF_UT  + 32768;    // U^T   [128][64]  bf16
static constexpr size_t OFF_A   = OFF_UBT + 16384;    // A     [64][64]   f32
static constexpr size_t OFF_BB  = OFF_A   + 16384;    // Bmat' [64][64]   bf16
static constexpr size_t OFF_LAM = OFF_BB  + 8192;
static constexpr size_t OFF_BET = OFF_LAM + 256;
static constexpr size_t OFF_RS  = OFF_BET + 256;
static constexpr size_t OFF_IL  = OFF_RS  + 256;
static constexpr size_t OFF_MET = OFF_IL  + 256;
static constexpr size_t DELTA_SMEM = OFF_MET + 64;    // 287808 B (< 320 KB)

__global__ void __launch_bounds__(256)
delta_kernel(const bf16_t* __restrict__ qi,  // [B, TI, H, DK]
             const bf16_t* __restrict__ ki,  // [B, TI, H, DK]
             const bf16_t* __restrict__ vi,  // [B, TI, H*DV]
             const float* __restrict__ gIn,  // [B, TI, H]
             const float* __restrict__ bIn,  // [B, TI, H]
             float* __restrict__ osel) {     // [B, T, H, DV]
    extern __shared__ char smem[];
    bf16_t* sST  = (bf16_t*)(smem + OFF_ST);
    bf16_t* sK   = (bf16_t*)(smem + OFF_K);
    bf16_t* sQ   = (bf16_t*)(smem + OFF_Q);
    bf16_t* sKi  = (bf16_t*)(smem + OFF_KI);
    bf16_t* sKiT = (bf16_t*)(smem + OFF_KIT);
    bf16_t* sVT  = (bf16_t*)(smem + OFF_VT);
    float*  sUT  = (float*) (smem + OFF_UT);
    bf16_t* sUbT = (bf16_t*)(smem + OFF_UBT);
    float*  sA   = (float*) (smem + OFF_A);
    bf16_t* sBb  = (bf16_t*)(smem + OFF_BB);
    float*  sLam  = (float*)(smem + OFF_LAM);
    float*  sBeta = (float*)(smem + OFF_BET);
    float*  sRS   = (float*)(smem + OFF_RS);
    float*  sInvL = (float*)(smem + OFF_IL);
    float*  sMeta = (float*)(smem + OFF_MET);

    int idx = (int)blockIdx.x;
    int sl = idx & (NSL - 1);
    int h  = (idx >> 2) % Hh;
    int bq = idx / (NSL * Hh);

    int tid = (int)threadIdx.x;
    int lane = tid & 31;
    int w = tid >> 5;

    // zero state
    for (int i = tid; i < DKc * DVS; i += 256) sST[i] = (bf16_t)0.f;
    __syncthreads();

    const bf16_t* kBase0 = ki + (((size_t)bq * TI) * Hh + h) * DKc;
    const bf16_t* qBase0 = qi + (((size_t)bq * TI) * Hh + h) * DKc;
    const bf16_t* vBase0 = vi + ((size_t)bq * TI) * VALD + h * DVc + sl * DVS;

    for (int cc = 0; cc < NCHUNK; ++cc) {
        int t0 = cc * CHKl;

        // ---- load per-step scalars (raw g into sLam) ----
        if (tid < 64) {
            size_t gb = ((size_t)bq * TI + (t0 + tid)) * Hh + h;
            sBeta[tid] = bIn[gb];
            sLam[tid]  = gIn[gb];
        }
        // ---- stage K/Q chunk tiles (64x256 bf16) via async-to-LDS ----
        const bf16_t* kb = kBase0 + (size_t)t0 * (Hh * DKc);
        const bf16_t* qb = qBase0 + (size_t)t0 * (Hh * DKc);
#pragma unroll
        for (int it = 0; it < 8; ++it) {
            int vid = tid + it * 256;       // 0..2047
            int row = vid >> 5;
            int vc  = (vid & 31) * 8;
            async_b128(sK + row * DKc + vc,
                       kb + (size_t)row * (Hh * DKc) + vc);
            if ((t0 + row) & 1) {           // q lives only at sub == NH-1
                async_b128(sQ + row * DKc + vc,
                           qb + (size_t)row * (Hh * DKc) + vc);
            } else {
                uint4 z = {0u, 0u, 0u, 0u};
                *(uint4*)(sQ + row * DKc + vc) = z;
            }
        }
        // ---- stage V^T via 8x8 register-block transpose (threads 0..127) ----
        const bf16_t* vb = vBase0 + (size_t)t0 * VALD;
        if (tid < 128) {
            int tb0 = (tid & 7) * 8;        // time base
            int dv0 = (tid >> 3) * 8;       // dv base
            PK8 ti[8], to[8];
#pragma unroll
            for (int e = 0; e < 8; ++e)
                ti[e].u = *(const uint4*)(vb + (size_t)(tb0 + e) * VALD + dv0);
#pragma unroll
            for (int f = 0; f < 8; ++f) {
#pragma unroll
                for (int e = 0; e < 8; ++e) to[f].h[e] = ti[e].h[f];
                *(uint4*)(sVT + (dv0 + f) * CHKl + tb0) = to[f].u;
            }
        }
        wait_async0();
        __syncthreads();

        // ---- cumulative log-decay + per-row scales ----
        if (tid == 0) {
            float a = 0.f;
            for (int i = 0; i < CHKl; ++i) { a += sLam[i]; sLam[i] = a; }
            sMeta[0] = a;
        }
        __syncthreads();
        if (tid < 64) {
            float bc = sLam[tid];
            float lam = expf(bc);
            sLam[tid]  = lam;               // Λ_i
            sInvL[tid] = expf(-bc);         // 1/Λ_i
            sRS[tid]   = sBeta[tid] * lam;  // β_i Λ_i
            if (tid == 0) sMeta[1] = expf(sMeta[0]);  // Λ_last
        }
        __syncthreads();

        // ---- stage Ki = (1/λ)K (row-major, vector scale copy) ----
#pragma unroll
        for (int it = 0; it < 8; ++it) {
            int vid = tid + it * 256;
            int row = vid >> 5;             // time j
            int vc  = (vid & 31) * 8;       // dk base
            PK8 t, o;
            t.u = *(const uint4*)(sK + row * DKc + vc);
            float sc = sInvL[row];
#pragma unroll
            for (int e = 0; e < 8; ++e) o.h[e] = (bf16_t)((float)t.h[e] * sc);
            *(uint4*)(sKi + row * DKc + vc) = o.u;
        }
        __syncthreads();

        // ---- stage Ki^T via 8x8 register-block transpose (all 256 threads) ----
        {
            int jb0 = (tid & 7) * 8;        // time base
            int dk0 = (tid >> 3) * 8;       // dk base
            PK8 ti[8], to[8];
#pragma unroll
            for (int e = 0; e < 8; ++e)
                ti[e].u = *(const uint4*)(sKi + (jb0 + e) * DKc + dk0);
#pragma unroll
            for (int f = 0; f < 8; ++f) {
#pragma unroll
                for (int e = 0; e < 8; ++e) to[f].h[e] = ti[e].h[f];
                *(uint4*)(sKiT + (dk0 + f) * CHKl + jb0) = to[f].u;
            }
        }
        __syncthreads();

        // ---- pass1: Amat, Bmat', and U-init (βV − βΛ·K@S0) ----
        for (int q8 = 0; q8 < 8; ++q8) {
            int jid = w * 8 + q8;
            v8f acc = v8f_zero();
            int half = lane >> 4, nn = lane & 15;
            if (jid < 16) {
                // Amat tile: raw K @ (Ki)^T, rows scaled by βΛ at store
                int tm = jid >> 2, tn = jid & 3;
                for (int k0 = 0; k0 < DKc; k0 += 32)
                    acc = wmma_bf16(frag_a_vec(sK, tm * 16, k0, DKc),
                                    frag_b_vec(sKi, k0, tn * 16, DKc), acc);
#pragma unroll
                for (int r = 0; r < 8; ++r) {
                    int i = tm * 16 + half * 8 + r, j = tn * 16 + nn;
                    sA[i * 64 + j] = (j < i) ? sRS[i] * acc[r] : 0.f;
                }
            } else if (jid < 32) {
                // Bmat' tile: raw Q @ (Ki)^T (λ_i row factor deferred to O)
                int j2 = jid - 16, tm = j2 >> 2, tn = j2 & 3;
                for (int k0 = 0; k0 < DKc; k0 += 32)
                    acc = wmma_bf16(frag_a_vec(sQ, tm * 16, k0, DKc),
                                    frag_b_vec(sKi, k0, tn * 16, DKc), acc);
#pragma unroll
                for (int r = 0; r < 8; ++r) {
                    int i = tm * 16 + half * 8 + r, j = tn * 16 + nn;
                    sBb[i * 64 + j] = (bf16_t)((j <= i) ? acc[r] : 0.f);
                }
            } else {
                // R tile: raw K @ S0; U-init rows scaled at store
                int j2 = jid - 32, tm = j2 >> 3, tn = j2 & 7;
                for (int k0 = 0; k0 < DKc; k0 += 32)
                    acc = wmma_bf16(frag_a_vec(sK, tm * 16, k0, DKc),
                                    frag_b_vec(sST, k0, tn * 16, DKc), acc);
                int i0 = tm * 16 + half * 8;
                int n  = tn * 16 + nn;
                PK8 tv;
                tv.u = *(const uint4*)(sVT + n * CHKl + i0);
                float* ucol = sUT + n * CHKl + i0;
#pragma unroll
                for (int r = 0; r < 8; ++r)
                    ucol[r] = sBeta[i0 + r] * (float)tv.h[r] -
                              sRS[i0 + r] * acc[r];
            }
        }
        __syncthreads();

        // ---- forward substitution: (I+A) U = P, column-parallel ----
        if (tid < DVS) {
            float* ucol = sUT + tid * CHKl;
            for (int i = 1; i < CHKl; ++i) {
                float s = 0.f;
                for (int j = 0; j < i; ++j) s += sA[i * 64 + j] * ucol[j];
                ucol[i] -= s;
            }
            bf16_t* ubcol = sUbT + tid * CHKl;
#pragma unroll
            for (int ib = 0; ib < 8; ++ib) {
                PK8 o;
#pragma unroll
                for (int e = 0; e < 8; ++e)
                    o.h[e] = (bf16_t)ucol[ib * 8 + e];
                *(uint4*)(ubcol + ib * 8) = o.u;
            }
        }
        __syncthreads();

        // ---- O = λ_i [ Q@S0 + Bmat'@U ]; store odd (q) rows ----
        for (int q4 = 0; q4 < 4; ++q4) {
            int jid = w * 4 + q4;
            int tm = jid >> 3, tn = jid & 7;
            v8f acc = v8f_zero();
            for (int k0 = 0; k0 < DKc; k0 += 32)
                acc = wmma_bf16(frag_a_vec(sQ, tm * 16, k0, DKc),
                                frag_b_vec(sST, k0, tn * 16, DKc), acc);
#pragma unroll
            for (int k0 = 0; k0 < CHKl; k0 += 32)
                acc = wmma_bf16(frag_a_vec(sBb, tm * 16, k0, CHKl),
                                frag_b_vec(sUbT, k0, tn * 16, CHKl), acc);
            int half = lane >> 4, nn = lane & 15;
#pragma unroll
            for (int r = 0; r < 8; ++r) {
                int i = tm * 16 + half * 8 + r;
                int tg = t0 + i;
                if (tg & 1) {
                    int t = tg >> 1;
                    osel[(((size_t)bq * Tlen + t) * Hh + h) * DVc + sl * DVS +
                         tn * 16 + nn] = sLam[i] * acc[r];
                }
            }
        }
        __syncthreads();  // all S0 reads complete before update

        // ---- S <- λ_last * ( S0 + (Ki)^T @ U ) ----
        float lamLast = sMeta[1];
        for (int q16 = 0; q16 < 16; ++q16) {
            int jid = w * 16 + q16;
            int tm = jid >> 3, tn = jid & 7;
            v8f acc = load_cT(sST, tm * 16, tn * 16, DKc);
#pragma unroll
            for (int k0 = 0; k0 < CHKl; k0 += 32)
                acc = wmma_bf16(frag_a_vec(sKiT, tm * 16, k0, CHKl),
                                frag_b_vec(sUbT, k0, tn * 16, CHKl), acc);
            store_cT(sST, tm * 16, tn * 16, DKc, acc, lamLast);
        }
        __syncthreads();
    }
}

// ---------------------------------------------------------------------------
// RMS norm over DV with o_norm_w, gate by silu(gate), write bf16 Y
// ---------------------------------------------------------------------------
__global__ void __launch_bounds__(256)
gate_rms_kernel(const float* __restrict__ osel,  // [B, T, H, DV]
                const float* __restrict__ gproj, // [B, T, VALD]
                const float* __restrict__ onw,   // [DV]
                bf16_t* __restrict__ Y) {        // [B, T, VALD]
    int blk = (int)blockIdx.x;
    int h = blk % Hh;
    int t = (blk / Hh) % Tlen;
    int b = blk / (Hh * Tlen);
    int tid = (int)threadIdx.x;

    const float* o = osel + (((size_t)b * Tlen + t) * Hh + h) * DVc;
    float o0 = o[tid], o1 = o[tid + 256];

    __shared__ float red[256];
    red[tid] = o0 * o0 + o1 * o1;
    __syncthreads();
#pragma unroll
    for (int s = 128; s > 0; s >>= 1) {
        if (tid < s) red[tid] += red[tid + s];
        __syncthreads();
    }
    float rn = rsqrtf(red[0] / (float)DVc + 1e-5f);

    size_t rowo = (size_t)(b * Tlen + t) * VALD + h * DVc;
#pragma unroll
    for (int half = 0; half < 2; ++half) {
        int dv = tid + half * 256;
        float ov = (half == 0) ? o0 : o1;
        float y = ov * rn * onw[dv];
        float gt = gproj[rowo + dv];
        y *= gt / (1.f + expf(-gt));  // silu(gate)
        Y[rowo + dv] = (bf16_t)y;
    }
}

// ---------------------------------------------------------------------------
// Host launcher
// ---------------------------------------------------------------------------
extern "C" void kernel_launch(void* const* d_in, const int* in_sizes, int n_in,
                              void* d_out, int out_size, void* d_ws,
                              size_t ws_size, hipStream_t stream) {
    (void)in_sizes; (void)n_in; (void)out_size; (void)ws_size;

    const float* x    = (const float*)d_in[0];
    const float* qw   = (const float*)d_in[1];
    const float* kws  = (const float*)d_in[2];
    const float* vws  = (const float*)d_in[3];
    const float* bws  = (const float*)d_in[4];
    const float* aw   = (const float*)d_in[5];
    const float* gw   = (const float*)d_in[6];
    const float* ow   = (const float*)d_in[7];
    const float* qcw  = (const float*)d_in[8];
    const float* kcw  = (const float*)d_in[9];
    const float* vcw  = (const float*)d_in[10];
    const float* Alog = (const float*)d_in[11];
    const float* dtb  = (const float*)d_in[12];
    const float* onw  = (const float*)d_in[13];
    float* out = (float*)d_out;

    char* ws = (char*)d_ws;
    size_t off = 0;
    auto take = [&](size_t bytes) -> char* {
        char* p = ws + off;
        off += (bytes + 255) & ~(size_t)255;
        return p;
    };

    const int M = Bsz * Tlen;  // 2048 rows

    bf16_t* xb  = (bf16_t*)take((size_t)M * HIDn * 2);
    bf16_t* wqb = (bf16_t*)take((size_t)HIDn * KEYD * 2);   // [N][K] transposed
    bf16_t* wk0 = (bf16_t*)take((size_t)HIDn * KEYD * 2);
    bf16_t* wk1 = (bf16_t*)take((size_t)HIDn * KEYD * 2);
    bf16_t* wv0 = (bf16_t*)take((size_t)HIDn * VALD * 2);
    bf16_t* wv1 = (bf16_t*)take((size_t)HIDn * VALD * 2);
    bf16_t* wgb = (bf16_t*)take((size_t)HIDn * VALD * 2);
    bf16_t* wob = (bf16_t*)take((size_t)VALD * HIDn * 2);
    float* qraw  = (float*)take((size_t)M * KEYD * 4);
    float* k0raw = (float*)take((size_t)M * KEYD * 4);
    float* k1raw = (float*)take((size_t)M * KEYD * 4);
    float* v0raw = (float*)take((size_t)M * VALD * 4);
    float* v1raw = (float*)take((size_t)M * VALD * 4);
    float* ggate = (float*)take((size_t)M * VALD * 4);
    bf16_t* qiB = (bf16_t*)take((size_t)Bsz * TI * KEYD * 2);
    bf16_t* kiB = (bf16_t*)take((size_t)Bsz * TI * KEYD * 2);
    bf16_t* viB = (bf16_t*)take((size_t)Bsz * TI * VALD * 2);
    float* giB = (float*)take((size_t)Bsz * TI * Hh * 4);
    float* biB = (float*)take((size_t)Bsz * TI * Hh * 4);
    float* osel = (float*)take((size_t)M * VALD * 4);
    bf16_t* Yb  = (bf16_t*)take((size_t)M * VALD * 2);

    // 1) casts: activations plain; weights cast + transpose to [N][K]
    cast_f32_to_bf16<<<2048, 256, 0, stream>>>(x, xb, (long)M * HIDn);
    cast_transpose_f32_to_bf16<<<2048, 256, 0, stream>>>(qw, wqb, HIDn, KEYD);
    cast_transpose_f32_to_bf16<<<2048, 256, 0, stream>>>(kws, wk0, HIDn, KEYD);
    cast_transpose_f32_to_bf16<<<2048, 256, 0, stream>>>(
        kws + (size_t)HIDn * KEYD, wk1, HIDn, KEYD);
    cast_transpose_f32_to_bf16<<<2048, 256, 0, stream>>>(vws, wv0, HIDn, VALD);
    cast_transpose_f32_to_bf16<<<2048, 256, 0, stream>>>(
        vws + (size_t)HIDn * VALD, wv1, HIDn, VALD);
    cast_transpose_f32_to_bf16<<<2048, 256, 0, stream>>>(gw, wgb, HIDn, VALD);
    cast_transpose_f32_to_bf16<<<2048, 256, 0, stream>>>(ow, wob, VALD, HIDn);

    // 2) projection GEMMs (WMMA bf16, f32 accumulate)
    dim3 blk(256);
    gemm_bf16<<<dim3(KEYD / 128, M / 128), blk, 0, stream>>>(xb, wqb, qraw, M, KEYD, HIDn);
    gemm_bf16<<<dim3(KEYD / 128, M / 128), blk, 0, stream>>>(xb, wk0, k0raw, M, KEYD, HIDn);
    gemm_bf16<<<dim3(KEYD / 128, M / 128), blk, 0, stream>>>(xb, wk1, k1raw, M, KEYD, HIDn);
    gemm_bf16<<<dim3(VALD / 128, M / 128), blk, 0, stream>>>(xb, wv0, v0raw, M, VALD, HIDn);
    gemm_bf16<<<dim3(VALD / 128, M / 128), blk, 0, stream>>>(xb, wv1, v1raw, M, VALD, HIDn);
    gemm_bf16<<<dim3(VALD / 128, M / 128), blk, 0, stream>>>(xb, wgb, ggate, M, VALD, HIDn);

    // 3) conv + silu + l2norm, interleave (q scale = DK^-0.5 = 1/16)
    conv_norm_qk<<<Bsz * Tlen * Hh, 256, 0, stream>>>(qraw, qcw, qiB, NHp - 1, 0.0625f);
    conv_norm_qk<<<Bsz * Tlen * Hh, 256, 0, stream>>>(k0raw, kcw, kiB, 0, 1.0f);
    conv_norm_qk<<<Bsz * Tlen * Hh, 256, 0, stream>>>(k1raw, kcw + (size_t)KEYD * 4, kiB, 1, 1.0f);
    conv_silu_v<<<Bsz * Tlen, 256, 0, stream>>>(v0raw, vcw, viB, 0);
    conv_silu_v<<<Bsz * Tlen, 256, 0, stream>>>(v1raw, vcw + (size_t)VALD * 4, viB, 1);

    // 4) beta / g scalars
    beta_g_kernel<<<Bsz * Tlen, 256, 0, stream>>>(x, bws, aw, Alog, dtb, biB, giB);

    // 5) chunked gated delta rule (48 blocks, ~281 KB dynamic LDS each)
    hipFuncSetAttribute(reinterpret_cast<const void*>(delta_kernel),
                        hipFuncAttributeMaxDynamicSharedMemorySize,
                        (int)DELTA_SMEM);
    delta_kernel<<<Bsz * Hh * NSL, 256, DELTA_SMEM, stream>>>(qiB, kiB, viB,
                                                              giB, biB, osel);

    // 6) RMS norm + gate
    gate_rms_kernel<<<Bsz * Tlen * Hh, 256, 0, stream>>>(osel, ggate, onw, Yb);

    // 7) output projection
    gemm_bf16<<<dim3(HIDn / 128, M / 128), blk, 0, stream>>>(Yb, wob, out, M,
                                                             HIDn, VALD);
}
